// SelfMatcher_53120155517067
// MI455X (gfx1250) — compile-verified, matrix operand order checked by reading, not run on hardware
//
#include <hip/hip_runtime.h>
#include <hip/hip_bf16.h>
#include <stdint.h>

// ---- problem dims (fixed by the reference) ----
#define L_SEQ 256
#define BATCH 64
#define DIM   512
#define D3    (3*DIM)

typedef __attribute__((ext_vector_type(16))) __bf16 v16bf;
typedef __attribute__((ext_vector_type(8)))  __bf16 v8bf;
typedef __attribute__((ext_vector_type(8)))  float  v8f;
typedef __attribute__((ext_vector_type(4)))  int    v4i;

// ---------- CDNA5 async global->LDS path (guarded; falls back cleanly) ----------
#if defined(__gfx1250__) && __has_builtin(__builtin_amdgcn_global_load_async_to_lds_b128)
#define HAVE_ASYNC_LDS 1
typedef __attribute__((address_space(1))) v4i glb_v4i;
typedef __attribute__((address_space(3))) v4i lds_v4i;
__device__ __forceinline__ void async_copy_b128(const void* gsrc, void* lds_dst) {
  __builtin_amdgcn_global_load_async_to_lds_b128(
      (glb_v4i*)(uintptr_t)gsrc,
      (lds_v4i*)(uint32_t)(uintptr_t)lds_dst, 0, 0);
}
#if __has_builtin(__builtin_amdgcn_s_wait_asynccnt)
#define WAIT_ASYNC(n) __builtin_amdgcn_s_wait_asynccnt(n)
#else
#define WAIT_ASYNC(n) asm volatile("s_wait_asynccnt %0" ::"i"(n) : "memory")
#endif
#else
#define HAVE_ASYNC_LDS 0
#endif

// ---------- fast transcendental helpers ----------
__device__ __forceinline__ float fast_exp2(float x) { return __builtin_amdgcn_exp2f(x); }
__device__ __forceinline__ float fast_rcp (float x) { return __builtin_amdgcn_rcpf(x); }
#if __has_builtin(__builtin_amdgcn_tanhf)
__device__ __forceinline__ float fast_tanh(float x) { return __builtin_amdgcn_tanhf(x); }
#else
__device__ __forceinline__ float fast_tanh(float x) {
  float e = fast_exp2(x * 2.8853900817779268f);  // exp(2x)
  return 1.0f - 2.0f * fast_rcp(e + 1.0f);
}
#endif
__device__ __forceinline__ float fast_sigmoid(float x) {
  return fast_rcp(1.0f + fast_exp2(-1.4426950408889634f * x));
}
__device__ __forceinline__ float wave_reduce_sum(float v) {
#pragma unroll
  for (int off = 16; off; off >>= 1) v += __shfl_xor(v, off, 32);
  return v;
}

// ---------- fp32 -> bf16 convert, 4 elements/thread ----------
__global__ __launch_bounds__(256) void cvt_f32_bf16x4(const float4* __restrict__ in,
                                                      uint2* __restrict__ out, int n4) {
  int i = blockIdx.x * 256 + threadIdx.x;
  if (i < n4) {
    float4 f = in[i];
    union { __bf16 b[4]; uint2 u; } pk;
    pk.b[0] = (__bf16)f.x; pk.b[1] = (__bf16)f.y;
    pk.b[2] = (__bf16)f.z; pk.b[3] = (__bf16)f.w;
    out[i] = pk.u;
  }
}

// ---------- v[l][b][d] (f32) -> vT[b][d][l] (bf16) ----------
__global__ __launch_bounds__(256) void transpose_v(const float* __restrict__ v,
                                                   __bf16* __restrict__ vT) {
  int idx = blockIdx.x * 256 + threadIdx.x;  // over B*D*L
  if (idx >= BATCH * DIM * L_SEQ) return;
  int l = idx % L_SEQ;
  int d = (idx / L_SEQ) % DIM;
  int b = idx / (L_SEQ * DIM);
  vT[idx] = (__bf16)v[((long long)l * BATCH + b) * DIM + d];
}

// ---------- WMMA fragment loader (16x32 bf16 from row-major [16][ld]) ----------
__device__ __forceinline__ v16bf load_frag(const __bf16* base, int ld, int k0) {
  int lane = threadIdx.x & 31;
  const __bf16* p = base + (lane & 15) * ld + k0 + ((lane >> 4) << 3);
  union { v16bf v; v8bf h[2]; } u;
  u.h[0] = *reinterpret_cast<const v8bf*>(p);
  u.h[1] = *reinterpret_cast<const v8bf*>(p + 16);
  return u.v;
}

// ---------- generic bf16 WMMA GEMM: C[m][n] = sum_k A[m][k]*B[n][k] (+bias[n]) ----------
// A: [M][K] bf16 row-major; B: [N][K] bf16 row-major. Block tile 128x128, K-chunk 32,
// 256 threads = 8 waves (2 M x 4 N), 4x2 WMMA tiles/wave. Double-buffered async LDS fill
// with the tail chunk peeled so the steady-state loop is branch-free.
__global__ __launch_bounds__(256)
void gemm_bf16(const __bf16* __restrict__ A, long long strideA,
               const __bf16* __restrict__ B, long long strideB,
               float* Cf, __bf16* Cb, long long strideC,
               int M, int N, int K, int ldc, const float* __restrict__ bias) {
  A += strideA * blockIdx.z;
  B += strideB * blockIdx.z;
  const int tid  = threadIdx.x;
  const int wid  = tid >> 5;
  const int wm   = wid & 1;
  const int wn   = wid >> 1;
  const int mBase = blockIdx.y * 128;
  const int nBase = blockIdx.x * 128;

  v8f vzero = {0.f, 0.f, 0.f, 0.f, 0.f, 0.f, 0.f, 0.f};
  v8f acc[4][2];
#pragma unroll
  for (int mt = 0; mt < 4; ++mt)
#pragma unroll
    for (int nt = 0; nt < 2; ++nt) acc[mt][nt] = vzero;

#if HAVE_ASYNC_LDS
  __shared__ __align__(16) __bf16 As[2][128 * 32];
  __shared__ __align__(16) __bf16 Bs[2][128 * 32];
  // per K-chunk each thread issues 4 async b128 copies (2 for A tile, 2 for B tile)
  const int u0r = tid >> 2,           u0c = (tid & 3) << 3;
  const int u1r = (tid + 256) >> 2,   u1c = ((tid + 256) & 3) << 3;

#define ISSUE_CHUNK(buf, kc)                                                           \
  do {                                                                                 \
    async_copy_b128(A + (long long)(mBase + u0r) * K + (kc) + u0c, &As[buf][tid * 8]); \
    async_copy_b128(B + (long long)(nBase + u0r) * K + (kc) + u0c, &Bs[buf][tid * 8]); \
    async_copy_b128(A + (long long)(mBase + u1r) * K + (kc) + u1c,                     \
                    &As[buf][(tid + 256) * 8]);                                        \
    async_copy_b128(B + (long long)(nBase + u1r) * K + (kc) + u1c,                     \
                    &Bs[buf][(tid + 256) * 8]);                                        \
  } while (0)

  auto mma_step = [&](int buf) {
    v16bf af[4], bfv[2];
#pragma unroll
    for (int mt = 0; mt < 4; ++mt) af[mt]  = load_frag(&As[buf][(wm * 64 + mt * 16) * 32], 32, 0);
#pragma unroll
    for (int nt = 0; nt < 2; ++nt) bfv[nt] = load_frag(&Bs[buf][(wn * 32 + nt * 16) * 32], 32, 0);
#pragma unroll
    for (int mt = 0; mt < 4; ++mt)
#pragma unroll
      for (int nt = 0; nt < 2; ++nt)
        acc[mt][nt] = __builtin_amdgcn_wmma_f32_16x16x32_bf16(
            false, af[mt], false, bfv[nt], (short)0, acc[mt][nt], false, false);
  };

  ISSUE_CHUNK(0, 0);
  int cur = 0;
  // steady state: branch-free issue-next / wait / barrier / MMA
  for (int kc = 0; kc + 32 < K; kc += 32) {
    __syncthreads();                 // everyone finished reading buf cur^1
    ISSUE_CHUNK(cur ^ 1, kc + 32);
    WAIT_ASYNC(4);                   // cur's 4 asyncs landed; next 4 stay in flight
    __syncthreads();                 // all waves' current chunk landed
    mma_step(cur);
    cur ^= 1;
  }
  // peeled tail chunk
  WAIT_ASYNC(0);
  __syncthreads();
  mma_step(cur);
#undef ISSUE_CHUNK
#else
  __shared__ __align__(16) __bf16 As[128 * 32];
  __shared__ __align__(16) __bf16 Bs[128 * 32];
  for (int kc = 0; kc < K; kc += 32) {
    __syncthreads();
#pragma unroll
    for (int t = 0; t < 2; ++t) {
      int u   = tid + t * 256;
      int row = u >> 2;
      int c8  = (u & 3) << 3;
      reinterpret_cast<uint4*>(As)[u] =
          *reinterpret_cast<const uint4*>(A + (long long)(mBase + row) * K + kc + c8);
      reinterpret_cast<uint4*>(Bs)[u] =
          *reinterpret_cast<const uint4*>(B + (long long)(nBase + row) * K + kc + c8);
    }
    __syncthreads();
    v16bf af[4], bfv[2];
#pragma unroll
    for (int mt = 0; mt < 4; ++mt) af[mt]  = load_frag(As + (wm * 64 + mt * 16) * 32, 32, 0);
#pragma unroll
    for (int nt = 0; nt < 2; ++nt) bfv[nt] = load_frag(Bs + (wn * 32 + nt * 16) * 32, 32, 0);
#pragma unroll
    for (int mt = 0; mt < 4; ++mt)
#pragma unroll
      for (int nt = 0; nt < 2; ++nt)
        acc[mt][nt] = __builtin_amdgcn_wmma_f32_16x16x32_bf16(
            false, af[mt], false, bfv[nt], (short)0, acc[mt][nt], false, false);
  }
#endif

  // D layout: VGPR r -> M = tile + r + (lane>=16 ? 8 : 0), N = tile + (lane&15)
  const int lane  = tid & 31;
  const int colL  = lane & 15;
  const int rowHi = (lane >> 4) << 3;
#pragma unroll
  for (int mt = 0; mt < 4; ++mt)
#pragma unroll
    for (int nt = 0; nt < 2; ++nt) {
      int col = nBase + wn * 32 + nt * 16 + colL;
      float bv = bias ? bias[col] : 0.0f;
#pragma unroll
      for (int r = 0; r < 8; ++r) {
        int row = mBase + wm * 64 + mt * 16 + rowHi + r;
        float val = acc[mt][nt][r] + bv;
        long long off = strideC * blockIdx.z + (long long)row * ldc + col;
        if (Cf) Cf[off] = val;
        else    Cb[off] = (__bf16)val;
      }
    }
}

// ---------- attention scores: s[b][i][l] = sum_h tanh(WpvI[i,b,h]+WpvA[l,b,h]) * Vvec[b,h] ----------
// grid: (4 l-chunks, B), 256 threads. Register-resident: WpvA rows (8 llocs x 16 bf16/lane)
// and Vvec slice stay in VGPRs for all 256 steps. No LDS, no barriers.
__global__ __launch_bounds__(256)
void score_kernel(const __bf16* __restrict__ WpvA, const __bf16* __restrict__ WpvI,
                  const float* __restrict__ Vvec, float* __restrict__ S) {
  const int b   = blockIdx.y;
  const int lc  = blockIdx.x;
  const int tid = threadIdx.x;
  const int wid = tid >> 5, lane = tid & 31;
  const int h0  = lane * 16;  // each lane owns 16 contiguous h

  float vvr[16];
#pragma unroll
  for (int e = 0; e < 16; ++e) vvr[e] = Vvec[b * DIM + h0 + e];

  v8bf ra[8][2];
#pragma unroll
  for (int jj = 0; jj < 8; ++jj) {
    long long row = ((long long)(lc * 64 + wid * 8 + jj) * BATCH + b) * DIM + h0;
    ra[jj][0] = *reinterpret_cast<const v8bf*>(WpvA + row);
    ra[jj][1] = *reinterpret_cast<const v8bf*>(WpvA + row + 8);
  }

  for (int i = 0; i < L_SEQ; ++i) {
    long long wrow = ((long long)i * BATCH + b) * DIM + h0;
    v8bf w0 = *reinterpret_cast<const v8bf*>(WpvI + wrow);
    v8bf w1 = *reinterpret_cast<const v8bf*>(WpvI + wrow + 8);
    float wf[16];
#pragma unroll
    for (int e = 0; e < 8; ++e) { wf[e] = (float)w0[e]; wf[8 + e] = (float)w1[e]; }
#pragma unroll
    for (int jj = 0; jj < 8; ++jj) {
      float acc = 0.f;
#pragma unroll
      for (int e = 0; e < 8; ++e)
        acc += fast_tanh(wf[e] + (float)ra[jj][0][e]) * vvr[e];
#pragma unroll
      for (int e = 0; e < 8; ++e)
        acc += fast_tanh(wf[8 + e] + (float)ra[jj][1][e]) * vvr[8 + e];
      acc = wave_reduce_sum(acc);
      if (lane == 0)
        S[((long long)b * L_SEQ + i) * L_SEQ + lc * 64 + wid * 8 + jj] = acc;
    }
  }
}

// ---------- softmax over l (rows of 256), one wave per row, bf16 probs out ----------
__global__ __launch_bounds__(256)
void softmax_kernel(const float* __restrict__ S, __bf16* __restrict__ P) {
  int row  = blockIdx.x * 8 + (threadIdx.x >> 5);
  int lane = threadIdx.x & 31;
  const float* s = S + (long long)row * L_SEQ;
  float vals[8];
  float mx = -1e30f;
#pragma unroll
  for (int j = 0; j < 8; ++j) { vals[j] = s[lane + j * 32]; mx = fmaxf(mx, vals[j]); }
#pragma unroll
  for (int off = 16; off; off >>= 1) mx = fmaxf(mx, __shfl_xor(mx, off, 32));
  float sum = 0.f;
#pragma unroll
  for (int j = 0; j < 8; ++j) {
    vals[j] = fast_exp2(1.4426950408889634f * (vals[j] - mx));
    sum += vals[j];
  }
  sum = wave_reduce_sum(sum);
  float inv = fast_rcp(sum);
  __bf16* p = P + (long long)row * L_SEQ;
#pragma unroll
  for (int j = 0; j < 8; ++j) p[lane + j * 32] = (__bf16)(vals[j] * inv);
}

// ---------- one GRU time step (sequential part), fused gh-WMMA + pointwise ----------
__global__ __launch_bounds__(256)
void gru_step(const float* __restrict__ hIn, float* __restrict__ hOut,
              const __bf16* __restrict__ whh, const float* __restrict__ gi,
              const float* __restrict__ bhh, float* __restrict__ out, int step) {
  extern __shared__ char smem[];
  __bf16* hA  = (__bf16*)smem;                  // 64 x 512 bf16 (64 KB)
  float*  ghS = (float*)(smem + 64 * 512 * 2);  // 64 x 192 f32 (48 KB)
  const int tid  = threadIdx.x;
  const int dBlk = blockIdx.x;

  for (int e4 = tid * 4; e4 < 64 * 512; e4 += 1024) {
    float4 f = *reinterpret_cast<const float4*>(hIn + e4);
    union { __bf16 b[4]; uint2 u; } pk;
    pk.b[0] = (__bf16)f.x; pk.b[1] = (__bf16)f.y;
    pk.b[2] = (__bf16)f.z; pk.b[3] = (__bf16)f.w;
    *reinterpret_cast<uint2*>(hA + e4) = pk.u;
  }
  __syncthreads();

  const int wid = tid >> 5;
  const int wm  = wid & 3;   // M tile (16 rows; M=64)
  const int wn  = wid >> 2;  // N group: 6 N-tiles each

  v8f vzero = {0.f, 0.f, 0.f, 0.f, 0.f, 0.f, 0.f, 0.f};
  v8f acc[6];
#pragma unroll
  for (int nt = 0; nt < 6; ++nt) acc[nt] = vzero;

  for (int kc = 0; kc < DIM; kc += 32) {
    v16bf af = load_frag(hA + wm * 16 * DIM, DIM, kc);
#pragma unroll
    for (int nt = 0; nt < 6; ++nt) {
      int ncol = wn * 96 + nt * 16;
      int gate = ncol >> 6;
      int dl   = ncol & 63;
      long long gRow = (long long)(gate * DIM + dBlk * 64 + dl);
      v16bf bfv = load_frag(whh + gRow * DIM, DIM, kc);
      acc[nt] = __builtin_amdgcn_wmma_f32_16x16x32_bf16(
          false, af, false, bfv, (short)0, acc[nt], false, false);
    }
  }

  const int lane = tid & 31;
  const int colL = lane & 15, rowHi = (lane >> 4) << 3;
#pragma unroll
  for (int nt = 0; nt < 6; ++nt) {
    int ncol = wn * 96 + nt * 16 + colL;
#pragma unroll
    for (int r = 0; r < 8; ++r) {
      int row = wm * 16 + rowHi + r;
      ghS[row * 192 + ncol] = acc[nt][r];
    }
  }
  __syncthreads();

  for (int e = tid; e < 64 * 64; e += 256) {
    int b  = e >> 6, dl = e & 63;
    int d  = dBlk * 64 + dl;
    const float* g = gi + ((long long)step * BATCH + b) * D3;
    float rp  = g[d]       + ghS[b * 192 + dl]       + bhh[d];
    float zp  = g[DIM + d] + ghS[b * 192 + 64 + dl]  + bhh[DIM + d];
    float hnp = ghS[b * 192 + 128 + dl] + bhh[2 * DIM + d];
    float r = fast_sigmoid(rp);
    float z = fast_sigmoid(zp);
    float n = fast_tanh(g[2 * DIM + d] + r * hnp);
    float hp = hIn[b * DIM + d];
    float hv = (1.f - z) * n + z * hp;
    hOut[b * DIM + d] = hv;
    out[((long long)step * BATCH + b) * DIM + d] = hv;
  }
}

// =======================================================================
extern "C" void kernel_launch(void* const* d_in, const int* in_sizes, int n_in,
                              void* d_out, int out_size, void* d_ws, size_t ws_size,
                              hipStream_t stream) {
  const float* v    = (const float*)d_in[0];
  const float* h0   = (const float*)d_in[1];
  const float* Vvec = (const float*)d_in[2];
  const float* Wp   = (const float*)d_in[3];
  const float* Wp_  = (const float*)d_in[4];
  const float* wih  = (const float*)d_in[5];
  const float* whh  = (const float*)d_in[6];
  const float* bih  = (const float*)d_in[7];
  const float* bhh  = (const float*)d_in[8];
  float* out = (float*)d_out;

  char* ws = (char*)d_ws;
  size_t off = 0;
  auto alloc = [&](size_t bytes) -> char* {
    char* p = ws + off;
    off += (bytes + 255) & ~(size_t)255;
    return p;
  };
  __bf16* v_bf    = (__bf16*)alloc((size_t)L_SEQ * BATCH * DIM * 2);
  __bf16* Wp_bf   = (__bf16*)alloc((size_t)DIM * DIM * 2);
  __bf16* Wpp_bf  = (__bf16*)alloc((size_t)DIM * DIM * 2);
  __bf16* wih_bf  = (__bf16*)alloc((size_t)D3 * DIM * 2);
  __bf16* whh_bf  = (__bf16*)alloc((size_t)D3 * DIM * 2);
  __bf16* vT_bf   = (__bf16*)alloc((size_t)BATCH * DIM * L_SEQ * 2);
  __bf16* WpvA_bf = (__bf16*)alloc((size_t)L_SEQ * BATCH * DIM * 2);
  __bf16* WpvI_bf = (__bf16*)alloc((size_t)L_SEQ * BATCH * DIM * 2);
  float*  S       = (float*)alloc((size_t)BATCH * L_SEQ * L_SEQ * 4);
  __bf16* P_bf    = (__bf16*)alloc((size_t)BATCH * L_SEQ * L_SEQ * 2);
  __bf16* c_bf    = (__bf16*)alloc((size_t)L_SEQ * BATCH * DIM * 2);
  float*  gi      = (float*)alloc((size_t)L_SEQ * BATCH * D3 * 4);
  float*  hbuf0   = (float*)alloc((size_t)BATCH * DIM * 4);
  float*  hbuf1   = (float*)alloc((size_t)BATCH * DIM * 4);

  // 1) bf16 conversions (vectorized x4)
  int n4;
  n4 = L_SEQ * BATCH * DIM / 4;
  cvt_f32_bf16x4<<<(n4 + 255) / 256, 256, 0, stream>>>((const float4*)v, (uint2*)v_bf, n4);
  n4 = DIM * DIM / 4;
  cvt_f32_bf16x4<<<(n4 + 255) / 256, 256, 0, stream>>>((const float4*)Wp, (uint2*)Wp_bf, n4);
  cvt_f32_bf16x4<<<(n4 + 255) / 256, 256, 0, stream>>>((const float4*)Wp_, (uint2*)Wpp_bf, n4);
  n4 = D3 * DIM / 4;
  cvt_f32_bf16x4<<<(n4 + 255) / 256, 256, 0, stream>>>((const float4*)wih, (uint2*)wih_bf, n4);
  cvt_f32_bf16x4<<<(n4 + 255) / 256, 256, 0, stream>>>((const float4*)whh, (uint2*)whh_bf, n4);
  int n = BATCH * DIM * L_SEQ;
  transpose_v<<<(n + 255) / 256, 256, 0, stream>>>(v, vT_bf);

  // 2) WpvA = v @ Wp_.T ; WpvI = v @ Wp.T   (M=16384, N=512, K=512, bf16 out)
  {
    dim3 g(DIM / 128, (L_SEQ * BATCH) / 128, 1);
    gemm_bf16<<<g, 256, 0, stream>>>(v_bf, 0, Wpp_bf, 0, nullptr, WpvA_bf, 0,
                                     L_SEQ * BATCH, DIM, DIM, DIM, nullptr);
    gemm_bf16<<<g, 256, 0, stream>>>(v_bf, 0, Wp_bf, 0, nullptr, WpvI_bf, 0,
                                     L_SEQ * BATCH, DIM, DIM, DIM, nullptr);
  }

  // 3) attention scores + softmax
  {
    dim3 g(L_SEQ / 64, BATCH, 1);
    score_kernel<<<g, 256, 0, stream>>>(WpvA_bf, WpvI_bf, Vvec, S);
    softmax_kernel<<<(BATCH * L_SEQ) / 8, 256, 0, stream>>>(S, P_bf);
  }

  // 4) context: per-b GEMM c_b[i,d] = sum_l a_b[i,l] * vT_b[d,l]  -> c[(i*B+b)][d] bf16
  {
    dim3 g(DIM / 128, L_SEQ / 128, BATCH);
    gemm_bf16<<<g, 256, 0, stream>>>(P_bf, (long long)L_SEQ * L_SEQ,
                                     vT_bf, (long long)DIM * L_SEQ,
                                     nullptr, c_bf, (long long)DIM,
                                     L_SEQ, DIM, L_SEQ, BATCH * DIM, nullptr);
  }

  // 5) gi = c @ w_ih.T + b_ih  (M=16384, N=1536, K=512, f32 out)
  {
    dim3 g(D3 / 128, (L_SEQ * BATCH) / 128, 1);
    gemm_bf16<<<g, 256, 0, stream>>>(c_bf, 0, wih_bf, 0, gi, nullptr, 0,
                                     L_SEQ * BATCH, D3, DIM, D3, bih);
  }

  // 6) sequential GRU: 256 small fused WMMA steps, h double-buffered
  (void)hipMemcpyAsync(hbuf0, h0, (size_t)BATCH * DIM * 4,
                       hipMemcpyDeviceToDevice, stream);
  size_t lds_gru = 64 * 512 * 2 + 64 * 192 * 4;  // 112 KB (under 320 KB/WGP)
  for (int i = 0; i < L_SEQ; ++i) {
    float* hi = (i & 1) ? hbuf1 : hbuf0;
    float* ho = (i & 1) ? hbuf0 : hbuf1;
    gru_step<<<8, 256, lds_gru, stream>>>(hi, ho, whh_bf, gi, bhh, out, i);
  }
  (void)in_sizes; (void)n_in; (void)out_size; (void)ws_size;
}
